// MultiHeadAttention_14559939133944
// MI455X (gfx1250) — matrix operand order repaired and checked by reading, not compiled, not run on hardware
//
#include <hip/hip_runtime.h>
#include <hip/hip_bf16.h>
#include <math.h>

#define NB   8
#define TT   1024
#define CC   256
#define HH   8
#define DD   32
#define MAXD 16
#define NREL 33

typedef __attribute__((ext_vector_type(16))) _Float16 v16h;
typedef __attribute__((ext_vector_type(8)))  _Float16 v8h;
typedef __attribute__((ext_vector_type(4)))  _Float16 v4h;
typedef __attribute__((ext_vector_type(8)))  float    v8f;
typedef __attribute__((ext_vector_type(4)))  float    v4f;
typedef _Float16 half_t;

// ---- WMMA operand loaders (wave32, CDNA5 VGPR layouts), all 16B-vector LDS loads ----

// A matrix 16x32 f16 from row-major LDS [16][ldm] (ldm*2 bytes must be 16B multiple).
static __device__ inline v16h ld_A(const half_t* p, int lane, int ldm) {
  const half_t* row = p + (lane & 15) * ldm + ((lane >> 4) << 3);
  v8h lo = *(const v8h*)(row);
  v8h hi = *(const v8h*)(row + 16);
  v16h a;
#pragma unroll
  for (int e = 0; e < 8; ++e) { a[e] = lo[e]; a[e + 8] = hi[e]; }
  return a;
}

// B matrix 32x16: B[k][n] = M[n][k] where LDS M is [16 rows(n)][ldm cols(k)] row-major.
static __device__ inline v16h ld_BT(const half_t* p, int lane, int ldm) {
  const half_t* row = p + (lane & 15) * ldm + ((lane >> 4) << 3);
  v8h lo = *(const v8h*)(row);
  v8h hi = *(const v8h*)(row + 16);
  v16h b;
#pragma unroll
  for (int e = 0; e < 8; ++e) { b[e] = lo[e]; b[e + 8] = hi[e]; }
  return b;
}

// ---- Kernel 0: transpose + convert a 256x256 weight to half: Wt[j][k] = (h)W[k][j] ----
__global__ __launch_bounds__(256)
void wprep_kernel(const float* __restrict__ W, half_t* __restrict__ Wt) {
  __shared__ float tl[32][33];
  const int bx = blockIdx.x * 32;            // k base
  const int by = blockIdx.y * 32;            // j base
  const int tx = threadIdx.x & 31, ty = threadIdx.x >> 5;   // ty 0..7
#pragma unroll
  for (int s = 0; s < 32; s += 8)
    tl[ty + s][tx] = W[(bx + ty + s) * CC + by + tx];       // coalesced over j
  __syncthreads();
#pragma unroll
  for (int s = 0; s < 32; s += 8)
    Wt[(by + ty + s) * CC + bx + tx] = (half_t)tl[tx][ty + s];  // coalesced over k
}

// ---- Kernel 1: project 33-row relative-position table through Wr ----
__global__ __launch_bounds__(256)
void reltab_kernel(const float* __restrict__ Wr, const float* __restrict__ br,
                   float* __restrict__ ptab) {
  __shared__ float ape[NREL][CC];
  const int j = threadIdx.x;
  const float log_inc = logf(10000.0f) / (float)(CC / 2 - 1);
  const float invf = __expf(-(float)(j & 127) * log_inc);
  for (int p = 0; p < NREL; ++p) {
    float st = (float)p * invf;
    ape[p][j] = (j < 128) ? __sinf(st) : __cosf(st);
  }
  __syncthreads();
  for (int p = 0; p < NREL; ++p) {
    float acc = br[j];
    for (int k = 0; k < CC; ++k) acc += ape[p][k] * Wr[k * CC + j];
    ptab[p * CC + j] = acc;
  }
}

// ---- Kernel 2: projection GEMM. NOUT 1|2; TRANS: store d-major [n][h][d][t] ----
template <int NOUT, bool TRANS>
__global__ __launch_bounds__(256)
void proj_kernel(const float* __restrict__ X, const half_t* __restrict__ Wt,
                 const float* __restrict__ bias,
                 const float* __restrict__ pe0, const float* __restrict__ pe1,
                 half_t* __restrict__ out0, half_t* __restrict__ out1) {
  __shared__ __align__(16) half_t At[64][32];
  __shared__ __align__(16) half_t BtT[64][40];   // W^T tile: [col][k], padded stride
  __shared__ __align__(16) half_t Ot[64][64];    // output staging

  const int tid  = threadIdx.x;
  const int lane = tid & 31;
  const int w    = tid >> 5;
  const int wm   = w & 3;
  const int wn   = w >> 2;
  const int m0   = blockIdx.x * 64;
  const int n0   = blockIdx.y * 64;

  v8f c0 = {}; v8f c1 = {};
  for (int k0 = 0; k0 < CC; k0 += 32) {
#pragma unroll
    for (int u = tid; u < 512; u += 256) {        // stage A: f32 v4 loads -> v4h stores
      const int row = u >> 3, cs = (u & 7) * 4;
      v4f x = *(const v4f*)&X[(m0 + row) * CC + k0 + cs];
      v4h hx;
#pragma unroll
      for (int j = 0; j < 4; ++j) hx[j] = (half_t)x[j];
      *(v4h*)&At[row][cs] = hx;
    }
    {                                             // stage B^T: 16B -> 16B
      const int c = tid >> 2, rs = (tid & 3) * 8; // 64 cols x 4 units
      *(v8h*)&BtT[c][rs] = *(const v8h*)&Wt[(n0 + c) * CC + k0 + rs];
    }
    __syncthreads();
    v16h a  = ld_A(&At[wm * 16][0], lane, 32);
    v16h b0 = ld_BT(&BtT[wn * 32][0], lane, 40);
    v16h b1 = ld_BT(&BtT[wn * 32 + 16][0], lane, 40);
    c0 = __builtin_amdgcn_wmma_f32_16x16x32_f16(false, a, false, b0, (short)0, c0, false, false);
    c1 = __builtin_amdgcn_wmma_f32_16x16x32_f16(false, a, false, b1, (short)0, c1, false, false);
    __syncthreads();
  }
  const int rl0 = wm * 16 + ((lane >> 4) << 3);
  const int cl0 = wn * 32 + (lane & 15);
  float vbuf[2][8];
#pragma unroll
  for (int t = 0; t < 2; ++t) {
    v8f& c = t ? c1 : c0;
    const float bc = bias[n0 + cl0 + t * 16];
#pragma unroll
    for (int r = 0; r < 8; ++r) vbuf[t][r] = c[r] + bc;
  }
#pragma unroll
  for (int p = 0; p < NOUT; ++p) {
    const float* pe = (p == 0) ? pe0 : pe1;
    half_t* dst = (p == 0) ? out0 : out1;
    __syncthreads();
#pragma unroll
    for (int t = 0; t < 2; ++t) {
      const int col = cl0 + t * 16;
      const float pv = (NOUT == 2) ? pe[n0 + col] : 0.f;
#pragma unroll
      for (int r = 0; r < 8; ++r) Ot[rl0 + r][col] = (half_t)(vbuf[t][r] + pv);
    }
    __syncthreads();
    if (!TRANS) {
#pragma unroll
      for (int u = tid; u < 512; u += 256) {      // row-major coalesced 16B stores
        const int row = u >> 3, cu = u & 7;
        *(v8h*)&dst[(size_t)(m0 + row) * CC + n0 + cu * 8] = *(const v8h*)&Ot[row][cu * 8];
      }
    } else {
#pragma unroll
      for (int u = tid; u < 512; u += 256) {      // d-major: [n][h][d][t], 16B over t
        const int cl = u >> 3, rseg = u & 7;
        v8h v;
#pragma unroll
        for (int j = 0; j < 8; ++j) v[j] = Ot[rseg * 8 + j][cl];
        const int gc = n0 + cl;                   // global channel
        const int hh = gc >> 5, dd = gc & 31;
        const int nn = m0 >> 10;                  // row tile never crosses batch
        const int tb = (m0 & 1023) + rseg * 8;
        *(v8h*)&dst[(((size_t)nn * HH + hh) * DD + dd) * TT + tb] = v;
      }
    }
  }
}

// ---- Kernel 3: S[n,h,t,j] = dot(q_v[n,h,t,:], ptab[j, h*D:(h+1)*D]) ----
__global__ __launch_bounds__(256)
void bd_kernel(const half_t* __restrict__ qvh, const float* __restrict__ ptab,
               float* __restrict__ S) {
  __shared__ float pt[NREL][DD];
  const int tid = threadIdx.x;
  const int h = blockIdx.y, n = blockIdx.z;
  for (int i = tid; i < NREL * DD; i += 256) {
    int p = i / DD, d = i % DD;
    pt[p][d] = ptab[p * CC + h * DD + d];
  }
  __syncthreads();
  const int t = blockIdx.x * 256 + tid;
  const half_t* qrow = &qvh[(size_t)(n * TT + t) * CC + h * DD];
  v8h q8[4];
#pragma unroll
  for (int u = 0; u < 4; ++u) q8[u] = *(const v8h*)&qrow[u * 8];
  float q[DD];
#pragma unroll
  for (int d = 0; d < DD; ++d) q[d] = (float)q8[d >> 3][d & 7];
  float* out = &S[(((size_t)n * HH + h) * TT + t) * NREL];
  for (int p = 0; p < NREL; ++p) {
    float acc = 0.f;
#pragma unroll
    for (int d = 0; d < DD; ++d) acc += q[d] * pt[p][d];
    out[p] = acc;
  }
}

// ---- Kernel 4: flash attention per (n, h, 64 q-rows) ----
__global__ __launch_bounds__(256)
void attn_kernel(const half_t* __restrict__ quh, const half_t* __restrict__ kh,
                 const half_t* __restrict__ vhT, const float* __restrict__ S,
                 const unsigned char* __restrict__ mask, half_t* __restrict__ aoh) {
  __shared__ __align__(16) half_t Qu[64][DD];      // also reused as output staging
  __shared__ __align__(16) half_t Kt[64][DD];      // [key][d]  (ld_BT gives K^T)
  __shared__ __align__(16) half_t VtT[DD][72];     // V^T: [d][key], padded stride
  __shared__ __align__(16) float  Sc[64][64];
  __shared__ __align__(16) half_t Pt[64][64];
  __shared__ float  rowM[64], rowL[64], rowScale[64];

  const int tid = threadIdx.x;
  const int lane = tid & 31;
  const int w = tid >> 5;
  const int q0 = blockIdx.x * 64;
  const int h  = blockIdx.y;
  const int n  = blockIdx.z;
  const float inv_sqrt_d = 0.17677669529663687f;

  {                                                // stage Q: one 16B unit per thread
    const int row = tid >> 2, cu = tid & 3;
    *(v8h*)&Qu[row][cu * 8] =
        *(const v8h*)&quh[(size_t)(n * TT + q0 + row) * CC + h * DD + cu * 8];
  }
  if (tid < 64) { rowM[tid] = -1e30f; rowL[tid] = 0.f; }
  v8f o = {};
  const int orow = (w & 3) * 16;
  const int ocol = (w >> 2) * 16;
  __syncthreads();

  const float* Sbase = &S[(((size_t)n * HH + h) * TT) * NREL];
  const half_t* vbase = &vhT[((size_t)n * HH + h) * DD * TT];

  for (int k0 = 0; k0 < TT; k0 += 64) {
    {                                              // stage K row-major: 16B units
      const int row = tid >> 2, cu = tid & 3;
      *(v8h*)&Kt[row][cu * 8] =
          *(const v8h*)&kh[(size_t)(n * TT + k0 + row) * CC + h * DD + cu * 8];
    }
    {                                              // stage V^T (d-major global): 16B units
      const int d = tid >> 3, seg = tid & 7;
      *(v8h*)&VtT[d][seg * 8] = *(const v8h*)&vbase[(size_t)d * TT + k0 + seg * 8];
    }
    __syncthreads();
    // scores: 16 (16x16) tiles over 64x64, 2 per wave, K-dim = D = 32 in one WMMA
#pragma unroll
    for (int tix = 0; tix < 2; ++tix) {
      const int tile = w * 2 + tix;
      const int qi = tile >> 2, ki = tile & 3;
      v16h a = ld_A(&Qu[qi * 16][0], lane, DD);
      v16h b = ld_BT(&Kt[ki * 16][0], lane, DD);
      v8f c = {};
      c = __builtin_amdgcn_wmma_f32_16x16x32_f16(false, a, false, b, (short)0, c, false, false);
      const int row0 = qi * 16 + ((lane >> 4) << 3);
      const int col  = ki * 16 + (lane & 15);
      const int kg = k0 + col;
      const bool mk = mask[n * TT + kg] != 0;
#pragma unroll
      for (int r = 0; r < 8; ++r) {
        const int row = row0 + r;
        const int qg = q0 + row;
        int d = kg - qg;
        d = (d < -MAXD) ? -MAXD : (d > MAXD ? MAXD : d);
        const float s = (c[r] + Sbase[(size_t)qg * NREL + d + MAXD]) * inv_sqrt_d;
        Sc[row][col] = mk ? s : -1e9f;
      }
    }
    __syncthreads();
    // online softmax: 4 threads per row
    {
      const int row = tid >> 2, sub = tid & 3;
      float lmax = -1e30f;
#pragma unroll
      for (int j = 0; j < 16; ++j) lmax = fmaxf(lmax, Sc[row][sub * 16 + j]);
      lmax = fmaxf(lmax, __shfl_xor(lmax, 1));
      lmax = fmaxf(lmax, __shfl_xor(lmax, 2));
      const float mold = rowM[row];
      const float mnew = fmaxf(mold, lmax);
      float lsum = 0.f;
#pragma unroll
      for (int j = 0; j < 16; ++j) {
        const float p = __expf(Sc[row][sub * 16 + j] - mnew);
        Pt[row][sub * 16 + j] = (half_t)p;
        lsum += p;
      }
      lsum += __shfl_xor(lsum, 1);
      lsum += __shfl_xor(lsum, 2);
      if (sub == 0) {
        const float sc = __expf(mold - mnew);
        rowScale[row] = sc;
        rowL[row] = rowL[row] * sc + lsum;
        rowM[row] = mnew;
      }
    }
    __syncthreads();
    // rescale running O and accumulate P (64x64) @ V (64x32)
    {
      const int rb = orow + ((lane >> 4) << 3);
#pragma unroll
      for (int r = 0; r < 8; ++r) o[r] *= rowScale[rb + r];
      v16h a0 = ld_A(&Pt[orow][0],  lane, 64);
      v16h a1 = ld_A(&Pt[orow][32], lane, 64);
      v16h b0 = ld_BT(&VtT[ocol][0],  lane, 72);
      v16h b1 = ld_BT(&VtT[ocol][32], lane, 72);
      o = __builtin_amdgcn_wmma_f32_16x16x32_f16(false, a0, false, b0, (short)0, o, false, false);
      o = __builtin_amdgcn_wmma_f32_16x16x32_f16(false, a1, false, b1, (short)0, o, false, false);
    }
    __syncthreads();
  }
  // normalize into LDS (reuse Qu), then coalesced 16B stores
  {
    const int rb = orow + ((lane >> 4) << 3);
    const int cl = ocol + (lane & 15);
#pragma unroll
    for (int r = 0; r < 8; ++r)
      Qu[rb + r][cl] = (half_t)(o[r] / rowL[rb + r]);
  }
  __syncthreads();
  {
    const int row = tid >> 2, cu = tid & 3;
    *(v8h*)&aoh[(size_t)(n * TT + q0 + row) * CC + h * DD + cu * 8] =
        *(const v8h*)&Qu[row][cu * 8];
  }
}

// ---- Kernel 5: output projection + bias + residual + layernorm ----
__global__ __launch_bounds__(256)
void out_kernel(const half_t* __restrict__ aoh, const half_t* __restrict__ WoT,
                const float* __restrict__ bo, const float* __restrict__ resid,
                const float* __restrict__ gamma, const float* __restrict__ beta,
                float* __restrict__ out) {
  __shared__ __align__(16) half_t At[32][32];
  __shared__ __align__(16) half_t BtT[256][40];   // Wo^T tile: [col][k]
  __shared__ __align__(16) float  Ob[32][256];

  const int tid = threadIdx.x;
  const int lane = tid & 31;
  const int w  = tid >> 5;
  const int wm = w & 1;
  const int wn = w >> 1;
  const int m0 = blockIdx.x * 32;

  v8f acc[4] = {};
  for (int k0 = 0; k0 < CC; k0 += 32) {
    if (tid < 128) {                               // stage A: 16B units
      const int row = tid >> 2, cu = tid & 3;
      *(v8h*)&At[row][cu * 8] =
          *(const v8h*)&aoh[(size_t)(m0 + row) * CC + k0 + cu * 8];
    }
#pragma unroll
    for (int u = tid; u < 1024; u += 256) {        // stage B^T: 16B units
      const int c = u >> 2, rs = (u & 3) * 8;
      *(v8h*)&BtT[c][rs] = *(const v8h*)&WoT[(size_t)c * CC + k0 + rs];
    }
    __syncthreads();
    v16h a = ld_A(&At[wm * 16][0], lane, 32);
#pragma unroll
    for (int t = 0; t < 4; ++t) {
      v16h b = ld_BT(&BtT[wn * 64 + t * 16][0], lane, 40);
      acc[t] = __builtin_amdgcn_wmma_f32_16x16x32_f16(false, a, false, b, (short)0, acc[t], false, false);
    }
    __syncthreads();
  }
  {
    const int rb = wm * 16 + ((lane >> 4) << 3);
#pragma unroll
    for (int t = 0; t < 4; ++t) {
      const int col = wn * 64 + t * 16 + (lane & 15);
      const float bc = bo[col];
#pragma unroll
      for (int r = 0; r < 8; ++r) {
        const int row = rb + r;
        Ob[row][col] = acc[t][r] + bc + resid[(m0 + row) * CC + col];
      }
    }
  }
  __syncthreads();
  // layernorm: 8 threads per row
  {
    const int row = tid >> 3, sub = tid & 7;
    float s = 0.f, s2 = 0.f;
#pragma unroll
    for (int j = 0; j < 32; ++j) {
      const float x = Ob[row][sub * 32 + j];
      s += x; s2 += x * x;
    }
    s  += __shfl_xor(s, 1);  s2 += __shfl_xor(s2, 1);
    s  += __shfl_xor(s, 2);  s2 += __shfl_xor(s2, 2);
    s  += __shfl_xor(s, 4);  s2 += __shfl_xor(s2, 4);
    const float mean = s * (1.f / 256.f);
    const float var  = s2 * (1.f / 256.f) - mean * mean;
    const float inv  = rsqrtf(var + 1e-6f);
#pragma unroll
    for (int j = 0; j < 32; ++j) {
      const int col = sub * 32 + j;
      const float x = Ob[row][col];
      out[(size_t)(m0 + row) * CC + col] = gamma[col] * ((x - mean) * inv) + beta[col];
    }
  }
}

extern "C" void kernel_launch(void* const* d_in, const int* in_sizes, int n_in,
                              void* d_out, int out_size, void* d_ws, size_t ws_size,
                              hipStream_t stream) {
  (void)in_sizes; (void)n_in; (void)out_size; (void)ws_size;
  const float* queries = (const float*)d_in[0];
  const float* keys    = (const float*)d_in[1];
  const unsigned char* key_mask = (const unsigned char*)d_in[2];
  const float* Wq = (const float*)d_in[3];
  const float* bq = (const float*)d_in[4];
  const float* Wk = (const float*)d_in[5];
  const float* bk = (const float*)d_in[6];
  const float* Wv = (const float*)d_in[7];
  const float* bv = (const float*)d_in[8];
  const float* Wo = (const float*)d_in[9];
  const float* bo = (const float*)d_in[10];
  const float* pe_u = (const float*)d_in[11];
  const float* pe_v = (const float*)d_in[12];
  const float* Wr = (const float*)d_in[13];
  const float* br = (const float*)d_in[14];
  const float* gamma = (const float*)d_in[15];
  const float* beta  = (const float*)d_in[16];

  char* ws = (char*)d_ws;
  const size_t szh = (size_t)NB * TT * CC * sizeof(half_t);   // 4 MiB each
  const size_t szS = ((size_t)NB * HH * TT * NREL * sizeof(float) + 255) & ~(size_t)255;
  const size_t szW = (size_t)CC * CC * sizeof(half_t);        // 128 KiB each
  half_t* quh = (half_t*)(ws);
  half_t* qvh = (half_t*)(ws + 1 * szh);
  half_t* khp = (half_t*)(ws + 2 * szh);
  half_t* vhT = (half_t*)(ws + 3 * szh);
  half_t* aoh = (half_t*)(ws + 4 * szh);
  float*  ptab = (float*)(ws + 5 * szh);
  float*  S    = (float*)(ws + 5 * szh + 33 * 1024);          // ptab pad to 33KB
  char*   wbase = ws + 5 * szh + 33 * 1024 + szS;
  half_t* WqT = (half_t*)(wbase + 0 * szW);
  half_t* WkT = (half_t*)(wbase + 1 * szW);
  half_t* WvT = (half_t*)(wbase + 2 * szW);
  half_t* WoT = (half_t*)(wbase + 3 * szW);

  const dim3 wg(CC / 32, CC / 32, 1);
  wprep_kernel<<<wg, 256, 0, stream>>>(Wq, WqT);
  wprep_kernel<<<wg, 256, 0, stream>>>(Wk, WkT);
  wprep_kernel<<<wg, 256, 0, stream>>>(Wv, WvT);
  wprep_kernel<<<wg, 256, 0, stream>>>(Wo, WoT);
  reltab_kernel<<<1, 256, 0, stream>>>(Wr, br, ptab);

  const dim3 pg((NB * TT) / 64, CC / 64, 1);
  proj_kernel<2, false><<<pg, 256, 0, stream>>>(queries, WqT, bq, pe_u, pe_v, quh, qvh);
  proj_kernel<1, false><<<pg, 256, 0, stream>>>(keys, WkT, bk, nullptr, nullptr, khp, nullptr);
  proj_kernel<1, true ><<<pg, 256, 0, stream>>>(keys, WvT, bv, nullptr, nullptr, vhT, nullptr);
  bd_kernel<<<dim3(TT / 256, HH, NB), 256, 0, stream>>>(qvh, ptab, S);
  attn_kernel<<<dim3(TT / 64, HH, NB), 256, 0, stream>>>(quh, khp, vhT, S, key_mask, aoh);
  out_kernel<<<dim3((NB * TT) / 32), 256, 0, stream>>>(aoh, WoT, bo, queries, gamma, beta,
                                                       (float*)d_out);
}
